// GCNEncoder_53145925320869
// MI455X (gfx1250) — matrix operand order repaired and checked by reading, not compiled, not run on hardware
//
#include <hip/hip_runtime.h>

#define CH 128

typedef __bf16 v16bf __attribute__((ext_vector_type(16)));
typedef float  v8f   __attribute__((ext_vector_type(8)));

union VB { v16bf v; unsigned u[8]; };

__device__ __forceinline__ unsigned short f2bf(float f) {
  unsigned u = __float_as_uint(f);
  u += 0x7FFFu + ((u >> 16) & 1u);          // round-to-nearest-even
  return (unsigned short)(u >> 16);
}
__device__ __forceinline__ float bf2f(unsigned short h) {
  return __uint_as_float(((unsigned)h) << 16);
}
__device__ __forceinline__ void split2(float f0, float f1, unsigned& hi, unsigned& lo) {
  unsigned short h0 = f2bf(f0), h1 = f2bf(f1);
  unsigned short l0 = f2bf(f0 - bf2f(h0)), l1 = f2bf(f1 - bf2f(h1));
  hi = (unsigned)h0 | ((unsigned)h1 << 16);
  lo = (unsigned)l0 | ((unsigned)l1 << 16);
}

// ---- K1: deg = 1 (self loop), zero stats accumulators -----------------------
__global__ __launch_bounds__(256) void k_init(float* deg, float* sums, int N) {
  int i = blockIdx.x * 256 + threadIdx.x;
  if (i < N)   deg[i]  = 1.0f;
  if (i < 256) sums[i] = 0.0f;     // sum[128] ++ sumsq[128]
}

// ---- K2: degree count over edges -------------------------------------------
__global__ __launch_bounds__(256) void k_edge_deg(const long long* __restrict__ ei,
                                                  float* deg, int E) {
  int i = blockIdx.x * 256 + threadIdx.x;
  if (i < E) atomicAdd(&deg[(int)ei[(size_t)E + i]], 1.0f);
}

// ---- K3: dinv = rsqrt(deg) in place ----------------------------------------
__global__ __launch_bounds__(256) void k_rsqrt(float* deg, int N) {
  int i = blockIdx.x * 256 + threadIdx.x;
  if (i < N) deg[i] = rsqrtf(deg[i]);
}

// ---- K4: hs = (x @ W^T) * dinv[row]; agg(d_out) seeded with self-loop term --
// bf16 split-product WMMA: h ~= xhi*Whi + xhi*Wlo + xlo*Whi (f32-class accuracy)
__global__ __launch_bounds__(256) void k_gemm_wmma(
    const float* __restrict__ x, const float* __restrict__ W,
    const float* __restrict__ dinv, float* __restrict__ hs,
    float* __restrict__ agg, int N)
{
  __shared__ unsigned sWhi[CH * CH / 2];   // packed bf16 pairs, 32 KB
  __shared__ unsigned sWlo[CH * CH / 2];   // residual plane,   32 KB
  const int t = threadIdx.x;
  for (int p = t; p < CH * CH / 2; p += 256)
    split2(W[2 * p], W[2 * p + 1], sWhi[p], sWlo[p]);
  __syncthreads();

  const int lane = t & 31, wave = t >> 5;
  const int tile = blockIdx.x * 8 + wave;
  const int numTiles = (N + 15) >> 4;
  if (tile >= numTiles) return;            // uniform per-wave branch, after barrier

  const int half = lane >> 4;              // 0: lanes 0-15, 1: lanes 16-31
  const int lcol = lane & 15;
  const int row0 = tile * 16;

  // A-matrix (16x32 bf16 per K-chunk), ISA zig layout: elem e=2j(+1):
  //   K = 32c + (j<4 ? 8*half+2j : 16+8*half+2(j-4)) (+1)
  const int m = (row0 + lcol < N) ? row0 + lcol : N - 1;
  const float* xr = x + (size_t)m * CH;
  VB aHi[4], aLo[4];
#pragma unroll
  for (int c = 0; c < 4; ++c) {
    const int kb = 32 * c + 8 * half;
    float tmp[16];
    const float4 r1 = *(const float4*)(xr + kb);
    const float4 r2 = *(const float4*)(xr + kb + 4);
    const float4 r3 = *(const float4*)(xr + kb + 16);
    const float4 r4 = *(const float4*)(xr + kb + 20);
    tmp[0]=r1.x;  tmp[1]=r1.y;  tmp[2]=r1.z;  tmp[3]=r1.w;
    tmp[4]=r2.x;  tmp[5]=r2.y;  tmp[6]=r2.z;  tmp[7]=r2.w;
    tmp[8]=r3.x;  tmp[9]=r3.y;  tmp[10]=r3.z; tmp[11]=r3.w;
    tmp[12]=r4.x; tmp[13]=r4.y; tmp[14]=r4.z; tmp[15]=r4.w;
#pragma unroll
    for (int j = 0; j < 8; ++j)
      split2(tmp[2 * j], tmp[2 * j + 1], aHi[c].u[j], aLo[c].u[j]);
  }

  // C/D layout: VGPR i -> row M = i + 8*half, col N = lcol
  float dv[8];
#pragma unroll
  for (int i = 0; i < 8; ++i) {
    int r = row0 + i + 8 * half;
    dv[i] = dinv[r < N ? r : N - 1];
  }

#pragma unroll
  for (int ot = 0; ot < 8; ++ot) {
    v8f acc;
#pragma unroll
    for (int i = 0; i < 8; ++i) acc[i] = 0.0f;
    const int o = ot * 16 + lcol;          // out-channel held by this lane
#pragma unroll
    for (int c = 0; c < 4; ++c) {
      // B-matrix (32x16 bf16): elem e -> K = 32c + (e<8 ? 8*half+e : 16+8*half+(e-8)), col = o
      VB bHi, bLo;
      const int base = o * 64 + 16 * c + 4 * half;
#pragma unroll
      for (int j = 0; j < 8; ++j) { bHi.u[j] = sWhi[base + j + (j & 4)]; bLo.u[j] = sWlo[base + j + (j & 4)]; }
      acc = __builtin_amdgcn_wmma_f32_16x16x32_bf16(false, aHi[c].v, false, bHi.v,
                                                    (short)0, acc, false, false);
      acc = __builtin_amdgcn_wmma_f32_16x16x32_bf16(false, aHi[c].v, false, bLo.v,
                                                    (short)0, acc, false, false);
      acc = __builtin_amdgcn_wmma_f32_16x16x32_bf16(false, aLo[c].v, false, bHi.v,
                                                    (short)0, acc, false, false);
    }
#pragma unroll
    for (int i = 0; i < 8; ++i) {
      const int r = row0 + i + 8 * half;
      if (r < N) {
        const float v = acc[i] * dv[i];
        const size_t idx = (size_t)r * CH + ot * 16 + lcol;
        hs[idx]  = v;   // scaled features for neighbors
        agg[idx] = v;   // self-loop seed (later *dinv[d] + b)
      }
    }
  }
}

// ---- K5: edge scatter: agg[dst] += hs[src]; one wave per edge ---------------
__global__ __launch_bounds__(256) void k_scatter(const long long* __restrict__ ei,
                                                 const float* __restrict__ hs,
                                                 float* __restrict__ agg, int E) {
  const long long gid = (long long)blockIdx.x * 256 + threadIdx.x;
  const long long eid = gid >> 5;
  if (eid >= E) return;
  const int lane = (int)(gid & 31);
  const int s = (int)ei[eid];
  const int d = (int)ei[(size_t)E + eid];
  const float4 v = *(const float4*)(hs + (size_t)s * CH + lane * 4);
  float* ap = agg + (size_t)d * CH + lane * 4;
  atomicAdd(ap + 0, v.x); atomicAdd(ap + 1, v.y);
  atomicAdd(ap + 2, v.z); atomicAdd(ap + 3, v.w);
}

// ---- K6: r = relu(dinv[n]*agg + b[c]) in place; accumulate BN stats ---------
__global__ __launch_bounds__(256) void k_relu_stats(
    const float* __restrict__ dinv, const float* __restrict__ b,
    float* __restrict__ rbuf, float* __restrict__ sum, float* __restrict__ sumsq, int N)
{
  __shared__ float sS[256], sQ[256];
  const int t = threadIdx.x, c = t & (CH - 1);
  const long long total = (long long)N * CH;
  const long long stride = (long long)gridDim.x * 256;   // multiple of 128 -> c fixed
  const float bc = b[c];
  float s = 0.0f, q = 0.0f;
  for (long long i = (long long)blockIdx.x * 256 + t; i < total; i += stride) {
    const int n = (int)(i >> 7);
    float v = dinv[n] * rbuf[i] + bc;
    v = v > 0.0f ? v : 0.0f;
    rbuf[i] = v;
    s += v; q += v * v;
  }
  sS[t] = s; sQ[t] = q;
  __syncthreads();
  if (t < 128) {
    atomicAdd(&sum[c],   sS[t] + sS[t + 128]);
    atomicAdd(&sumsq[c], sQ[t] + sQ[t + 128]);
  }
}

// ---- K7: finalize scale/shift ----------------------------------------------
__global__ void k_bn_finalize(const float* sum, const float* sumsq,
                              const float* gamma, const float* beta,
                              float* stats, int N) {
  const int c = threadIdx.x;
  if (c < CH) {
    const float mean = sum[c] / (float)N;
    const float var  = sumsq[c] / (float)N - mean * mean;
    const float sc   = gamma[c] * rsqrtf(var + 1e-5f);
    stats[c]      = sc;
    stats[CH + c] = beta[c] - mean * sc;
  }
}

// ---- K8: out = r * scale + shift, in place ---------------------------------
__global__ __launch_bounds__(256) void k_bn_apply(float* __restrict__ out,
                                                  const float* __restrict__ stats,
                                                  long long total) {
  const long long stride = (long long)gridDim.x * 256;
  for (long long i = (long long)blockIdx.x * 256 + threadIdx.x; i < total; i += stride) {
    const int c = (int)(i & (CH - 1));
    out[i] = out[i] * stats[c] + stats[CH + c];
  }
}

extern "C" void kernel_launch(void* const* d_in, const int* in_sizes, int n_in,
                              void* d_out, int out_size, void* d_ws, size_t ws_size,
                              hipStream_t stream) {
  const float*     x     = (const float*)d_in[0];
  const long long* ei    = (const long long*)d_in[1];   // int64 [2, E]
  const float*     W     = (const float*)d_in[2];
  const float*     b     = (const float*)d_in[3];
  const float*     gamma = (const float*)d_in[4];
  const float*     beta  = (const float*)d_in[5];
  const int N = in_sizes[0] / CH;
  const int E = in_sizes[1] / 2;

  float* out = (float*)d_out;                 // agg -> r -> final (in place)
  float* w   = (float*)d_ws;
  float* deg = w;                             // N floats; becomes dinv in place
  size_t off = ((size_t)N + 255) & ~(size_t)255;
  float* hs  = w + off;                       // N*CH floats
  off += (size_t)N * CH;
  float* sum   = w + off;                     // 128
  float* sumsq = sum + CH;                    // 128
  float* stats = sumsq + CH;                  // 256: scale | shift

  const int initGrid = ((N > 512 ? N : 512) + 255) / 256;
  k_init<<<initGrid, 256, 0, stream>>>(deg, sum, N);
  k_edge_deg<<<(E + 255) / 256, 256, 0, stream>>>(ei, deg, E);
  k_rsqrt<<<(N + 255) / 256, 256, 0, stream>>>(deg, N);

  const int numTiles = (N + 15) / 16;
  k_gemm_wmma<<<(numTiles + 7) / 8, 256, 0, stream>>>(x, W, deg, hs, out, N);

  k_scatter<<<(int)(((long long)E * 32 + 255) / 256), 256, 0, stream>>>(ei, hs, out, E);
  k_relu_stats<<<1024, 256, 0, stream>>>(deg, b, out, sum, sumsq, N);
  k_bn_finalize<<<1, 128, 0, stream>>>(sum, sumsq, gamma, beta, stats, N);
  k_bn_apply<<<4096, 256, 0, stream>>>(out, stats, (long long)N * CH);
}
